// TransformerModel_12936441495924
// MI455X (gfx1250) — compile-verified, matrix operand order checked by reading, not run on hardware
//
#include <hip/hip_runtime.h>
#include <cstdint>
#include <cstddef>

// Problem constants (match reference)
#define B_C   2
#define S_C   512
#define T_C   512
#define E_C   1024
#define NH_C  16
#define HD_C  64
#define HID_C 4096
#define L_C   6
#define V_C   32000

typedef __attribute__((ext_vector_type(16))) __bf16 v16bf;
typedef __attribute__((ext_vector_type(8)))  float  v8f;

__device__ __forceinline__ unsigned short f2bf_bits(float f) {
  union { float f; uint32_t u; } c; c.f = f;
  uint32_t u = c.u;
  u += 0x7FFFu + ((u >> 16) & 1u);   // round-to-nearest-even
  return (unsigned short)(u >> 16);
}

// ---------------------------------------------------------------------------
// Batched bf16 WMMA GEMM (B always given transposed):
//   C[M,N] = act( alpha * A[M,K] x Bt[N,K]^T + bias + Res )
// Both operands are loaded K-contiguous per lane -> global_load_b128 pairs.
// Each wave computes a 16x32 tile (one A fragment feeds two WMMAs); the
// wave-uniform "two" case is hoisted OUT of the K-loop so both loop bodies
// are branch-free (no per-iteration EXEC manipulation, full unroll+pipeline).
// Batch z decomposed (outer = z/innerB, inner = z%innerB) with independent
// strides per matrix -> handles [b,h] attention heads.
// transCb: write bf16 output transposed (Cb[col*ldcb + row]) - used to emit
// V^T directly from the V projection so P.V is also a fast-path GEMM.
// ---------------------------------------------------------------------------
__global__ void __launch_bounds__(256)
gemm_bf16_wmma(const unsigned short* __restrict__ A, long lda, long sAo, long sAi,
               const unsigned short* __restrict__ Bt, long ldb, long sBo, long sBi,
               const float* __restrict__ bias,
               const float* __restrict__ Res, long ldr, long sRo, long sRi,
               float* __restrict__ C, long ldc, long sCo, long sCi,
               unsigned short* __restrict__ Cb, long ldcb, long sCbo, long sCbi,
               int transCb,
               int M, int N, int K, int innerB, float alpha, int relu)
{
  const int wave = threadIdx.x >> 5;
  const int lane = threadIdx.x & 31;
  const int half = lane >> 4;       // K-group select (ISA 16-bit A layout)
  const int r    = lane & 15;       // row (A) / col (B,C)

  const int m0 = blockIdx.x * 16;
  const int n0 = (blockIdx.y * 8 + wave) * 32;
  if (m0 >= M || n0 >= N) return;            // wave-uniform
  const bool two = (n0 + 16) < N;            // wave-uniform

  const int zo = (int)blockIdx.z / innerB;
  const int zi = (int)blockIdx.z % innerB;

  const unsigned short* Ab = A  + (size_t)zo * sAo + (size_t)zi * sAi;
  const unsigned short* Bb = Bt + (size_t)zo * sBo + (size_t)zi * sBi;

  union Frag { v16bf v; uint32_t u[8]; };
  v8f acc0 = {}, acc1 = {};

  // Packed-pair k offsets per ISA 7.12.2 (16-bit A/B fragment layout):
  //   uint index within a 32-K block = (j&3) + ((j>>2)<<3) + (half<<2)
  const uint32_t* arow  = (const uint32_t*)(Ab + (size_t)(m0 + r) * lda);
  const uint32_t* brow0 = (const uint32_t*)(Bb + (size_t)(n0 + r) * ldb);
  const uint32_t* brow1 = (const uint32_t*)(Bb + (size_t)(n0 + 16 + r) * ldb);

  if (two) {
    for (int kk = 0; kk < K; kk += 32) {
      const int kb = kk >> 1;
      __builtin_prefetch((const char*)(arow  + kb) + 1024, 0, 3);
      __builtin_prefetch((const char*)(brow0 + kb) + 1024, 0, 3);
      __builtin_prefetch((const char*)(brow1 + kb) + 1024, 0, 3);
      Frag af, bf0, bf1;
#pragma unroll
      for (int j = 0; j < 8; ++j) {
        const int idx = kb + (j & 3) + ((j >> 2) << 3) + (half << 2);
        af.u[j]  = arow[idx];
        bf0.u[j] = brow0[idx];
        bf1.u[j] = brow1[idx];
      }
      acc0 = __builtin_amdgcn_wmma_f32_16x16x32_bf16(
          false, af.v, false, bf0.v, (short)0, acc0, false, false);
      acc1 = __builtin_amdgcn_wmma_f32_16x16x32_bf16(
          false, af.v, false, bf1.v, (short)0, acc1, false, false);
    }
  } else {
    for (int kk = 0; kk < K; kk += 32) {
      const int kb = kk >> 1;
      __builtin_prefetch((const char*)(arow  + kb) + 1024, 0, 3);
      __builtin_prefetch((const char*)(brow0 + kb) + 1024, 0, 3);
      Frag af, bf0;
#pragma unroll
      for (int j = 0; j < 8; ++j) {
        const int idx = kb + (j & 3) + ((j >> 2) << 3) + (half << 2);
        af.u[j]  = arow[idx];
        bf0.u[j] = brow0[idx];
      }
      acc0 = __builtin_amdgcn_wmma_f32_16x16x32_bf16(
          false, af.v, false, bf0.v, (short)0, acc0, false, false);
    }
  }

  // Epilogue: C/D layout (ISA): VGPR i -> row = i + 8*half, col = lane&15
#pragma unroll
  for (int t = 0; t < 2; ++t) {
    if (t == 1 && !two) break;
    const int col = n0 + t * 16 + r;
#pragma unroll
    for (int i = 0; i < 8; ++i) {
      const int row = m0 + i + 8 * half;
      float v = alpha * ((t == 0) ? acc0[i] : acc1[i]);
      if (bias) v += bias[col];
      if (Res)  v += Res[(size_t)zo * sRo + (size_t)zi * sRi + (size_t)row * ldr + col];
      if (relu) v = v > 0.f ? v : 0.f;
      if (C)  C [(size_t)zo * sCo  + (size_t)zi * sCi  + (size_t)row * ldc  + col] = v;
      if (Cb) {
        const size_t o = transCb
            ? ((size_t)zo * sCbo + (size_t)zi * sCbi + (size_t)col * ldcb + row)
            : ((size_t)zo * sCbo + (size_t)zi * sCbi + (size_t)row * ldcb + col);
        Cb[o] = f2bf_bits(v);
      }
    }
  }
}

// ---------------------------------------------------------------------------
// LDS-tiled transpose + fp32->bf16: in[K,N] -> out[N,K]. Grid (N/32, K/32, L).
// Makes every weight GEMM use the contiguous (b128) Bt fast path.
// ---------------------------------------------------------------------------
__global__ void __launch_bounds__(256)
transpose_f32_bf16(const float* __restrict__ in, unsigned short* __restrict__ out,
                   int K, int N, long sIn, long sOut)
{
  __shared__ unsigned short tile[32][33];
  const float* ip = in + (size_t)blockIdx.z * sIn;
  unsigned short* op = out + (size_t)blockIdx.z * sOut;
  const int n0 = blockIdx.x * 32;
  const int k0 = blockIdx.y * 32;
  const int tx = threadIdx.x & 31;
  const int ty = threadIdx.x >> 5;
#pragma unroll
  for (int p = 0; p < 4; ++p) {
    const int k = k0 + ty + p * 8;
    tile[ty + p * 8][tx] = f2bf_bits(ip[(size_t)k * N + n0 + tx]);
  }
  __syncthreads();
#pragma unroll
  for (int p = 0; p < 4; ++p) {
    const int n = n0 + ty + p * 8;
    op[(size_t)n * K + k0 + tx] = tile[tx][ty + p * 8];
  }
}

// ---------------------------------------------------------------------------
// LayerNorm over last dim, bf16 output (feeds WMMA GEMMs). One block per row.
// ---------------------------------------------------------------------------
__global__ void __launch_bounds__(256)
layernorm_bf16(const float* __restrict__ x, const float* __restrict__ g,
               const float* __restrict__ b, unsigned short* __restrict__ out, int E)
{
  __shared__ float red[256];
  const int row = blockIdx.x;
  const int tid = threadIdx.x;
  const float* xr = x + (size_t)row * E;

  float s = 0.f;
  for (int i = tid; i < E; i += 256) s += xr[i];
  red[tid] = s; __syncthreads();
  for (int o = 128; o > 0; o >>= 1) { if (tid < o) red[tid] += red[tid + o]; __syncthreads(); }
  const float mu = red[0] / (float)E;
  __syncthreads();

  float v = 0.f;
  for (int i = tid; i < E; i += 256) { float d = xr[i] - mu; v += d * d; }
  red[tid] = v; __syncthreads();
  for (int o = 128; o > 0; o >>= 1) { if (tid < o) red[tid] += red[tid + o]; __syncthreads(); }
  const float rstd = rsqrtf(red[0] / (float)E + 1e-5f);

  for (int i = tid; i < E; i += 256)
    out[(size_t)row * E + i] = f2bf_bits((xr[i] - mu) * rstd * g[i] + b[i]);
}

// ---------------------------------------------------------------------------
// Masked softmax over keys, bf16 probabilities out.
// mode 0: enc self (key pad && query pad, src) | mode 1: dec self (causal &&
// key pad, tgt) | mode 2: cross (key pad, src).  Fill = -10000.0 (reference).
// ---------------------------------------------------------------------------
__global__ void __launch_bounds__(256)
softmax_mask(const float* __restrict__ s, unsigned short* __restrict__ p,
             const int* __restrict__ qtok, const int* __restrict__ ktok,
             int SQ, int SK, int NH, int mode)
{
  __shared__ float red[256];
  const int row = blockIdx.x;              // b*NH*SQ + h*SQ + q
  const int tid = threadIdx.x;
  const int q = row % SQ;
  const int b = row / (NH * SQ);
  const float* sr = s + (size_t)row * SK;
  unsigned short* pr = p + (size_t)row * SK;

  const bool qv = (mode == 0) ? (qtok[(size_t)b * SQ + q] != 0) : true;

  float vals[4];
  int cnt = 0;
  float mx = -3.4e38f;
  for (int k = tid; k < SK; k += 256) {
    float v = sr[k];
    const bool kv = (ktok[(size_t)b * SK + k] != 0);
    bool ok;
    if (mode == 0)      ok = kv && qv;
    else if (mode == 1) ok = kv && (k <= q);
    else                ok = kv;
    if (!ok) v = -10000.0f;
    vals[cnt++] = v;
    mx = fmaxf(mx, v);
  }
  red[tid] = mx; __syncthreads();
  for (int o = 128; o > 0; o >>= 1) { if (tid < o) red[tid] = fmaxf(red[tid], red[tid + o]); __syncthreads(); }
  mx = red[0]; __syncthreads();

  float sum = 0.f;
  for (int i = 0; i < cnt; ++i) { vals[i] = __expf(vals[i] - mx); sum += vals[i]; }
  red[tid] = sum; __syncthreads();
  for (int o = 128; o > 0; o >>= 1) { if (tid < o) red[tid] += red[tid + o]; __syncthreads(); }
  const float inv = 1.f / red[0];

  cnt = 0;
  for (int k = tid; k < SK; k += 256) pr[k] = f2bf_bits(vals[cnt++] * inv);
}

// ---------------------------------------------------------------------------
// Embedding gather * sqrt(E) + sinusoidal positional encoding (inline)
// ---------------------------------------------------------------------------
__global__ void __launch_bounds__(256)
embed_pe(const int* __restrict__ tok, const float* __restrict__ emb,
         float* __restrict__ x, int S, int E)
{
  const int row = blockIdx.x;              // b*S + s
  const int pos = row % S;
  const int t = tok[row];
  const float* er = emb + (size_t)t * E;
  const float scale = sqrtf((float)E);
  const float c = -logf(10000.f) / (float)E;
  for (int e = threadIdx.x; e < E; e += 256) {
    const int i2 = e & ~1;
    const float ang = (float)pos * expf(c * (float)i2);
    const float pe = (e & 1) ? cosf(ang) : sinf(ang);
    x[(size_t)row * E + e] = er[e] * scale + pe;
  }
}

// fp32 -> bf16 (grid-stride, no transpose) - used for enc_out
__global__ void __launch_bounds__(256)
cvt_f32_bf16(const float* __restrict__ in, unsigned short* __restrict__ out, long n)
{
  long i = (long)blockIdx.x * 256 + threadIdx.x;
  const long stride = (long)gridDim.x * 256;
  for (; i < n; i += stride) out[i] = f2bf_bits(in[i]);
}

// ---------------------------------------------------------------------------
// Host-side helpers
// ---------------------------------------------------------------------------
static inline void launch_gemm(hipStream_t st,
    const unsigned short* A, long lda, long sAo, long sAi,
    const unsigned short* Bt, long ldb, long sBo, long sBi,
    const float* bias,
    const float* Res, long ldr, long sRo, long sRi,
    float* C, long ldc, long sCo, long sCi,
    unsigned short* Cb, long ldcb, long sCbo, long sCbi, int transCb,
    int M, int N, int K, int outerB, int innerB, float alpha, int relu)
{
  dim3 grid((unsigned)(M / 16), (unsigned)((N + 255) / 256), (unsigned)(outerB * innerB));
  gemm_bf16_wmma<<<grid, 256, 0, st>>>(A, lda, sAo, sAi, Bt, ldb, sBo, sBi,
                                       bias, Res, ldr, sRo, sRi, C, ldc, sCo, sCi,
                                       Cb, ldcb, sCbo, sCbi, transCb,
                                       M, N, K, innerB, alpha, relu);
}

extern "C" void kernel_launch(void* const* d_in, const int* in_sizes, int n_in,
                              void* d_out, int out_size, void* d_ws, size_t ws_size,
                              hipStream_t stream)
{
  (void)in_sizes; (void)n_in; (void)out_size; (void)ws_size;

  // ---- inputs (setup_inputs dict order) ----
  const int*   src      = (const int*)  d_in[0];
  const int*   tgt      = (const int*)  d_in[1];
  const float* emb_enc  = (const float*)d_in[2];
  const float* emb_dec  = (const float*)d_in[3];
  const float* enc_Wq   = (const float*)d_in[4];
  const float* enc_Wk   = (const float*)d_in[5];
  const float* enc_Wv   = (const float*)d_in[6];
  const float* enc_ffW1 = (const float*)d_in[7];
  const float* enc_ffb1 = (const float*)d_in[8];
  const float* enc_ffW2 = (const float*)d_in[9];
  const float* enc_ffb2 = (const float*)d_in[10];
  const float* enc_ln1g = (const float*)d_in[11];
  const float* enc_ln1b = (const float*)d_in[12];
  const float* enc_ln2g = (const float*)d_in[13];
  const float* enc_ln2b = (const float*)d_in[14];
  const float* dec_sWq  = (const float*)d_in[15];
  const float* dec_sWk  = (const float*)d_in[16];
  const float* dec_sWv  = (const float*)d_in[17];
  const float* dec_cWq  = (const float*)d_in[18];
  const float* dec_cWk  = (const float*)d_in[19];
  const float* dec_cWv  = (const float*)d_in[20];
  const float* dec_ffW1 = (const float*)d_in[21];
  const float* dec_ffb1 = (const float*)d_in[22];
  const float* dec_ffW2 = (const float*)d_in[23];
  const float* dec_ffb2 = (const float*)d_in[24];
  const float* dec_ln1g = (const float*)d_in[25];
  const float* dec_ln1b = (const float*)d_in[26];
  const float* dec_ln2g = (const float*)d_in[27];
  const float* dec_ln2b = (const float*)d_in[28];
  const float* dec_ln3g = (const float*)d_in[29];
  const float* dec_ln3b = (const float*)d_in[30];
  const float* fin_lng  = (const float*)d_in[31];
  const float* fin_lnb  = (const float*)d_in[32];
  const float* out_W    = (const float*)d_in[33];
  const float* out_b    = (const float*)d_in[34];

  float* logits = (float*)d_out;          // [B, T, V] fp32

  // ---- workspace carve-up (bump allocator, 256B aligned) ----
  char* wp = (char*)d_ws;
  auto alloc = [&](size_t bytes) -> void* {
    void* p = (void*)wp;
    wp += (bytes + 255) & ~(size_t)255;
    return p;
  };
  const size_t EE  = (size_t)E_C * E_C;
  const size_t EH  = (size_t)E_C * HID_C;
  const size_t BSE = (size_t)B_C * S_C * E_C;      // 1,048,576

  // bf16 transposed weights [N,K] (~366 MB)
  unsigned short* wEncWq = (unsigned short*)alloc(2 * L_C * EE);
  unsigned short* wEncWk = (unsigned short*)alloc(2 * L_C * EE);
  unsigned short* wEncWv = (unsigned short*)alloc(2 * L_C * EE);
  unsigned short* wEncW1 = (unsigned short*)alloc(2 * L_C * EH);
  unsigned short* wEncW2 = (unsigned short*)alloc(2 * L_C * EH);
  unsigned short* wDsWq  = (unsigned short*)alloc(2 * L_C * EE);
  unsigned short* wDsWk  = (unsigned short*)alloc(2 * L_C * EE);
  unsigned short* wDsWv  = (unsigned short*)alloc(2 * L_C * EE);
  unsigned short* wDcWq  = (unsigned short*)alloc(2 * L_C * EE);
  unsigned short* wDcWk  = (unsigned short*)alloc(2 * L_C * EE);
  unsigned short* wDcWv  = (unsigned short*)alloc(2 * L_C * EE);
  unsigned short* wDecW1 = (unsigned short*)alloc(2 * L_C * EH);
  unsigned short* wDecW2 = (unsigned short*)alloc(2 * L_C * EH);
  unsigned short* wOut   = (unsigned short*)alloc(2 * (size_t)E_C * V_C);

  // activations
  float*          x     = (float*)alloc(4 * BSE);                       // encoder stream
  float*          y     = (float*)alloc(4 * BSE);                       // decoder stream
  unsigned short* nbf   = (unsigned short*)alloc(2 * BSE);              // layernorm out
  unsigned short* qbf   = (unsigned short*)alloc(2 * BSE);              // [b,s,h*d]
  unsigned short* kbf   = (unsigned short*)alloc(2 * BSE);              // [b,s,h*d]
  unsigned short* vtb   = (unsigned short*)alloc(2 * BSE);              // V^T: [b][h*d][s]
  float*          scores= (float*)alloc(4 * (size_t)B_C * NH_C * S_C * S_C);
  unsigned short* pbf   = (unsigned short*)alloc(2 * (size_t)B_C * NH_C * S_C * S_C);
  unsigned short* h1bf  = (unsigned short*)alloc(2 * (size_t)B_C * S_C * HID_C);
  unsigned short* encbf = (unsigned short*)alloc(2 * BSE);              // bf16 enc_out

  // ---- 1) weight fp32->bf16 transpose (deterministic each call) ----
  auto tw = [&](const float* a, unsigned short* b, int K, int N, int Lcnt) {
    dim3 g((unsigned)(N / 32), (unsigned)(K / 32), (unsigned)Lcnt);
    transpose_f32_bf16<<<g, 256, 0, stream>>>(a, b, K, N, (long)K * N, (long)K * N);
  };
  tw(enc_Wq,   wEncWq, E_C,  E_C,  L_C);  tw(enc_Wk,   wEncWk, E_C,  E_C,  L_C);
  tw(enc_Wv,   wEncWv, E_C,  E_C,  L_C);
  tw(enc_ffW1, wEncW1, E_C,  HID_C, L_C); tw(enc_ffW2, wEncW2, HID_C, E_C, L_C);
  tw(dec_sWq,  wDsWq,  E_C,  E_C,  L_C);  tw(dec_sWk,  wDsWk,  E_C,  E_C,  L_C);
  tw(dec_sWv,  wDsWv,  E_C,  E_C,  L_C);
  tw(dec_cWq,  wDcWq,  E_C,  E_C,  L_C);  tw(dec_cWk,  wDcWk,  E_C,  E_C,  L_C);
  tw(dec_cWv,  wDcWv,  E_C,  E_C,  L_C);
  tw(dec_ffW1, wDecW1, E_C,  HID_C, L_C); tw(dec_ffW2, wDecW2, HID_C, E_C, L_C);
  tw(out_W,    wOut,   E_C,  V_C,  1);

  const int BS = B_C * S_C;                 // 1024 rows
  const long SE  = (long)S_C * E_C;
  const long ES  = (long)E_C * S_C;         // V^T per-batch stride
  const long HDS = (long)HD_C * S_C;        // V^T per-head stride
  const long HSS = (long)S_C * S_C;         // per-head score stride
  const long BSS = (long)NH_C * S_C * S_C;  // per-batch score stride
  const float iscale = 0.125f;              // 1/sqrt(HD)

  // ---- 2) embeddings + PE ----
  embed_pe<<<BS, 256, 0, stream>>>(src, emb_enc, x, S_C, E_C);
  embed_pe<<<BS, 256, 0, stream>>>(tgt, emb_dec, y, T_C, E_C);

  // ---- 3) encoder layers ----
  for (int i = 0; i < L_C; ++i) {
    const size_t iE = (size_t)i * E_C;
    layernorm_bf16<<<BS, 256, 0, stream>>>(x, enc_ln1g + iE, enc_ln1b + iE, nbf, E_C);
    launch_gemm(stream, nbf, E_C, 0, 0, wEncWq + i * EE, E_C, 0, 0, nullptr,
                nullptr, 0, 0, 0, nullptr, 0, 0, 0, qbf, E_C, 0, 0, 0,
                BS, E_C, E_C, 1, 1, 1.f, 0);
    launch_gemm(stream, nbf, E_C, 0, 0, wEncWk + i * EE, E_C, 0, 0, nullptr,
                nullptr, 0, 0, 0, nullptr, 0, 0, 0, kbf, E_C, 0, 0, 0,
                BS, E_C, E_C, 1, 1, 1.f, 0);
    // V projection writes V^T [b][h*d][s] (batched over b so index stays affine)
    launch_gemm(stream, nbf, E_C, SE, 0, wEncWv + i * EE, E_C, 0, 0, nullptr,
                nullptr, 0, 0, 0, nullptr, 0, 0, 0, vtb, S_C, ES, 0, 1,
                S_C, E_C, E_C, B_C, 1, 1.f, 0);
    // scores[b,h,q,k] = (Q.K^T)/sqrt(HD)
    launch_gemm(stream, qbf, E_C, SE, HD_C, kbf, E_C, SE, HD_C, nullptr,
                nullptr, 0, 0, 0, scores, S_C, BSS, HSS, nullptr, 0, 0, 0, 0,
                S_C, S_C, HD_C, B_C, NH_C, iscale, 0);
    softmax_mask<<<B_C * NH_C * S_C, 256, 0, stream>>>(scores, pbf, src, src, S_C, S_C, NH_C, 0);
    // x += merge_heads(P.V)  (Bt = V^T rows are d, length SK)
    launch_gemm(stream, pbf, S_C, BSS, HSS, vtb, S_C, ES, HDS, nullptr,
                x, E_C, SE, HD_C, x, E_C, SE, HD_C, nullptr, 0, 0, 0, 0,
                S_C, HD_C, S_C, B_C, NH_C, 1.f, 0);
    // FFN
    layernorm_bf16<<<BS, 256, 0, stream>>>(x, enc_ln2g + iE, enc_ln2b + iE, nbf, E_C);
    launch_gemm(stream, nbf, E_C, 0, 0, wEncW1 + i * EH, E_C, 0, 0,
                enc_ffb1 + (size_t)i * HID_C, nullptr, 0, 0, 0,
                nullptr, 0, 0, 0, h1bf, HID_C, 0, 0, 0, BS, HID_C, E_C, 1, 1, 1.f, 1);
    launch_gemm(stream, h1bf, HID_C, 0, 0, wEncW2 + i * EH, HID_C, 0, 0,
                enc_ffb2 + iE, x, E_C, 0, 0, x, E_C, 0, 0, nullptr, 0, 0, 0, 0,
                BS, E_C, HID_C, 1, 1, 1.f, 0);
  }
  cvt_f32_bf16<<<dim3(4096), 256, 0, stream>>>(x, encbf, (long)BSE);   // enc_out bf16

  // ---- 4) decoder layers ----
  for (int i = 0; i < L_C; ++i) {
    const size_t iE = (size_t)i * E_C;
    // self-attention (causal + tgt pad)
    layernorm_bf16<<<BS, 256, 0, stream>>>(y, dec_ln1g + iE, dec_ln1b + iE, nbf, E_C);
    launch_gemm(stream, nbf, E_C, 0, 0, wDsWq + i * EE, E_C, 0, 0, nullptr,
                nullptr, 0, 0, 0, nullptr, 0, 0, 0, qbf, E_C, 0, 0, 0, BS, E_C, E_C, 1, 1, 1.f, 0);
    launch_gemm(stream, nbf, E_C, 0, 0, wDsWk + i * EE, E_C, 0, 0, nullptr,
                nullptr, 0, 0, 0, nullptr, 0, 0, 0, kbf, E_C, 0, 0, 0, BS, E_C, E_C, 1, 1, 1.f, 0);
    launch_gemm(stream, nbf, E_C, SE, 0, wDsWv + i * EE, E_C, 0, 0, nullptr,
                nullptr, 0, 0, 0, nullptr, 0, 0, 0, vtb, T_C, ES, 0, 1,
                T_C, E_C, E_C, B_C, 1, 1.f, 0);
    launch_gemm(stream, qbf, E_C, SE, HD_C, kbf, E_C, SE, HD_C, nullptr,
                nullptr, 0, 0, 0, scores, T_C, BSS, HSS, nullptr, 0, 0, 0, 0,
                T_C, T_C, HD_C, B_C, NH_C, iscale, 0);
    softmax_mask<<<B_C * NH_C * T_C, 256, 0, stream>>>(scores, pbf, tgt, tgt, T_C, T_C, NH_C, 1);
    launch_gemm(stream, pbf, T_C, BSS, HSS, vtb, T_C, ES, HDS, nullptr,
                y, E_C, SE, HD_C, y, E_C, SE, HD_C, nullptr, 0, 0, 0, 0,
                T_C, HD_C, T_C, B_C, NH_C, 1.f, 0);
    // cross-attention (src pad on keys)
    layernorm_bf16<<<BS, 256, 0, stream>>>(y, dec_ln2g + iE, dec_ln2b + iE, nbf, E_C);
    launch_gemm(stream, nbf, E_C, 0, 0, wDcWq + i * EE, E_C, 0, 0, nullptr,
                nullptr, 0, 0, 0, nullptr, 0, 0, 0, qbf, E_C, 0, 0, 0, BS, E_C, E_C, 1, 1, 1.f, 0);
    launch_gemm(stream, encbf, E_C, 0, 0, wDcWk + i * EE, E_C, 0, 0, nullptr,
                nullptr, 0, 0, 0, nullptr, 0, 0, 0, kbf, E_C, 0, 0, 0, BS, E_C, E_C, 1, 1, 1.f, 0);
    launch_gemm(stream, encbf, E_C, SE, 0, wDcWv + i * EE, E_C, 0, 0, nullptr,
                nullptr, 0, 0, 0, nullptr, 0, 0, 0, vtb, S_C, ES, 0, 1,
                S_C, E_C, E_C, B_C, 1, 1.f, 0);
    launch_gemm(stream, qbf, E_C, SE, HD_C, kbf, E_C, SE, HD_C, nullptr,
                nullptr, 0, 0, 0, scores, S_C, BSS, HSS, nullptr, 0, 0, 0, 0,
                T_C, S_C, HD_C, B_C, NH_C, iscale, 0);
    softmax_mask<<<B_C * NH_C * T_C, 256, 0, stream>>>(scores, pbf, src, src, T_C, S_C, NH_C, 2);
    launch_gemm(stream, pbf, S_C, BSS, HSS, vtb, S_C, ES, HDS, nullptr,
                y, E_C, SE, HD_C, y, E_C, SE, HD_C, nullptr, 0, 0, 0, 0,
                T_C, HD_C, S_C, B_C, NH_C, 1.f, 0);
    // FFN
    layernorm_bf16<<<BS, 256, 0, stream>>>(y, dec_ln3g + iE, dec_ln3b + iE, nbf, E_C);
    launch_gemm(stream, nbf, E_C, 0, 0, wDecW1 + i * EH, E_C, 0, 0,
                dec_ffb1 + (size_t)i * HID_C, nullptr, 0, 0, 0,
                nullptr, 0, 0, 0, h1bf, HID_C, 0, 0, 0, BS, HID_C, E_C, 1, 1, 1.f, 1);
    launch_gemm(stream, h1bf, HID_C, 0, 0, wDecW2 + i * EH, HID_C, 0, 0,
                dec_ffb2 + iE, y, E_C, 0, 0, y, E_C, 0, 0, nullptr, 0, 0, 0, 0,
                BS, E_C, HID_C, 1, 1, 1.f, 0);
  }

  // ---- 5) final LN + output projection -> logits ----
  layernorm_bf16<<<BS, 256, 0, stream>>>(y, fin_lng, fin_lnb, nbf, E_C);
  launch_gemm(stream, nbf, E_C, 0, 0, wOut, E_C, 0, 0, out_b,
              nullptr, 0, 0, 0, logits, V_C, 0, 0, nullptr, 0, 0, 0, 0,
              BS, V_C, E_C, 1, 1, 1.f, 0);
}
